// MultiAgentGRU_45930380264230
// MI455X (gfx1250) — matrix operand order repaired and checked by reading, not compiled, not run on hardware
//
#include <hip/hip_runtime.h>
#include <hip/hip_bf16.h>

// Problem constants (from reference): B=256, T=128, A=4, I=128, H=256
#define Bv 256
#define Tv 128
#define Av 4
#define Iv 128
#define Hv 256
#define G3 (3 * Hv)   // 768 gate columns (r | z | n)

typedef __bf16 bf16_t;
typedef __attribute__((ext_vector_type(16))) __bf16 v16bf;
typedef __attribute__((ext_vector_type(8)))  __bf16 v8bf;
typedef __attribute__((ext_vector_type(8)))  float  v8f;

__device__ __forceinline__ v8f wmma_bf16(v16bf a, v16bf b, v8f c) {
  // D = A(16x32 bf16) * B(32x16 bf16) + C(16x16 f32)
  return __builtin_amdgcn_wmma_f32_16x16x32_bf16(false, a, false, b,
                                                 (short)0, c, false, false);
}

// ---------------------------------------------------------------------------
// Prep kernel: one-time f32 -> bf16 conversion of W_ih (768x128) and
// W_hh (768x256) into workspace. 576 KB total -> stays resident in the
// 192 MB L2 for the whole scan; every B-fragment read hits L2.
// ---------------------------------------------------------------------------
__global__ void cvt_weights(const float* __restrict__ wihf,
                            const float* __restrict__ whhf,
                            bf16_t* __restrict__ wih,
                            bf16_t* __restrict__ whh) {
  int i = blockIdx.x * blockDim.x + threadIdx.x;
  if (i < G3 * Iv) wih[i] = (bf16_t)wihf[i];
  if (i < G3 * Hv) whh[i] = (bf16_t)whhf[i];
}

// ---------------------------------------------------------------------------
// Fused GRU scan. One workgroup = 4 batches x 4 agents = 16 rows (one WMMA
// M-tile, no padding waste). 512 threads = 16 waves; wave w owns H-columns
// [16w, 16w+16). Per timestep per wave: 36 v_wmma_f32_16x16x32_bf16.
// B-fragments stream from the bf16 weight image (L2-resident) every step; a
// laundered zero OFFSET (not a laundered pointer) keeps the loads loop-variant
// (no LICM -> no 288-VGPR spill) while preserving the global address space so
// they lower to global_load_b128 (LOADcnt only), not flat_load (LOADcnt+DScnt).
// ---------------------------------------------------------------------------
#define XS 136   // x LDS row stride in bf16 (128 + pad, 272 B, 16B-aligned rows)
#define HS 264   // h LDS row stride (256 + pad, bf16 rows 528 B, f32 rows 1056 B)

__global__ __launch_bounds__(512)
void gru_scan(const float* __restrict__ x,
              const int*   __restrict__ is_init,      // (B,T,1), nonzero = reset h
              const bf16_t* __restrict__ wih_g,        // [768][128] bf16 row-major
              const bf16_t* __restrict__ whh_g,        // [768][256] bf16 row-major
              const float* __restrict__ b_ih,
              const float* __restrict__ b_hh,
              float* __restrict__ out) {
  __shared__ bf16_t xl[16 * XS];     // current x tile, bf16 (A-operand)
  __shared__ bf16_t hb[16 * HS];     // h state, bf16 (A-operand)
  __shared__ float  hf32[16 * HS];   // h state, f32 (exact z*h term)
  __shared__ float  bi[G3];
  __shared__ float  bh[G3];

  const int tid  = threadIdx.x;
  const int lane = tid & 31;
  const int wv   = tid >> 5;            // wave 0..15
  const int b0   = blockIdx.x * 4;      // first batch of this WG

  // biases -> LDS
  for (int i = tid; i < G3; i += 512) { bi[i] = b_ih[i]; bh[i] = b_hh[i]; }

  // h0 = 0
  {
    int m = tid >> 5, seg = (tid & 31) * 8;
    #pragma unroll
    for (int e = 0; e < 8; ++e) {
      hf32[m * HS + seg + e] = 0.f;
      hb  [m * HS + seg + e] = (bf16_t)0.f;
    }
  }

  // Documented gfx1250 WMMA lane mappings:
  const int ml    = lane & 15;          // A row / B column / D column
  const int khalf = (lane >> 4) * 8;    // unused (AMD table-order K offset 8h)
  const int kb    = (lane >> 4) * 16;   // A and B K base: k = 16h + s (this code's contiguous order)
  const int mb    = (lane >> 4) * 8;    // C/D row base (reg r -> row mb+r)
  const int hc    = wv * 16;            // this wave's H-column base

  float* hn_out = out + (size_t)Bv * Tv * Av * Hv;

  for (int t = 0; t < Tv; ++t) {
    __syncthreads();                    // (1) previous step's h writes visible

    // h = where(is_init[b,t], 0, h) — per-batch reset, broadcast over agents
    {
      int m = tid >> 5, seg = (tid & 31) * 8;
      int bg = b0 + (m >> 2);
      if (is_init[bg * Tv + t]) {
        #pragma unroll
        for (int e = 0; e < 8; ++e) {
          hf32[m * HS + seg + e] = 0.f;
          hb  [m * HS + seg + e] = (bf16_t)0.f;
        }
      }
    }
    // Stage x[b, t, a, :] tile (16 rows x 128) f32 -> bf16 LDS; prefetch t+1.
    {
      int m = tid >> 5, i0 = (tid & 31) * 4;
      int bg = b0 + (m >> 2), a = m & 3;
      const float* src = x + (size_t)((bg * Tv + t) * Av + a) * Iv + i0;
      float4 v = *(const float4*)src;
      bf16_t* dst = &xl[m * XS + i0];
      dst[0] = (bf16_t)v.x; dst[1] = (bf16_t)v.y;
      dst[2] = (bf16_t)v.z; dst[3] = (bf16_t)v.w;
      if (t + 1 < Tv) __builtin_prefetch(src + Av * Iv, 0, 3);  // next step's x
    }
    __syncthreads();                    // (2) staging + reset visible

    // ---- A-fragments from LDS (16-bit A 16x32 layout: two 16B chunks) ----
    v16bf xf[4], hfr[8];
    #pragma unroll
    for (int kt = 0; kt < 4; ++kt) {
      const bf16_t* p = &xl[ml * XS + kt * 32 + kb];
      v8bf lo = *(const v8bf*)p;
      v8bf hi = *(const v8bf*)(p + 8);
      #pragma unroll
      for (int e = 0; e < 8; ++e) { xf[kt][e] = lo[e]; xf[kt][e + 8] = hi[e]; }
    }
    #pragma unroll
    for (int kt = 0; kt < 8; ++kt) {
      const bf16_t* p = &hb[ml * HS + kt * 32 + kb];
      v8bf lo = *(const v8bf*)p;
      v8bf hi = *(const v8bf*)(p + 8);
      #pragma unroll
      for (int e = 0; e < 8; ++e) { hfr[kt][e] = lo[e]; hfr[kt][e + 8] = hi[e]; }
    }
    // old h (f32) for this wave's 16 columns
    float hold[8];
    #pragma unroll
    for (int r = 0; r < 8; ++r) hold[r] = hf32[(mb + r) * HS + hc + ml];
    __syncthreads();                    // (3) all reads of old h complete

    // Laundered zero offset: volatile asm (emits no code) makes the weight
    // addresses loop-variant (defeats LICM/spill) while keeping the GEP base
    // = kernel-arg pointer, so addrspace inference still yields global loads.
    unsigned long long zofs = 0;
    asm volatile("" : "+s"(zofs));
    const bf16_t* wih = wih_g + zofs;
    const bf16_t* whh = whh_g + zofs;

    // ---- gate GEMMs: B-fragments are contiguous 32B slices of weight rows ----
    v8f ar{}, az{}, anx{}, anh{};
    const int gr = hc, gz = Hv + hc, gn = 2 * Hv + hc;
    #pragma unroll
    for (int kt = 0; kt < 4; ++kt) {    // x-part, K = 128
      int ko = kt * 32 + kb;
      ar  = wmma_bf16(xf[kt], *(const v16bf*)(wih + (size_t)(gr + ml) * Iv + ko), ar);
      az  = wmma_bf16(xf[kt], *(const v16bf*)(wih + (size_t)(gz + ml) * Iv + ko), az);
      anx = wmma_bf16(xf[kt], *(const v16bf*)(wih + (size_t)(gn + ml) * Iv + ko), anx);
    }
    #pragma unroll
    for (int kt = 0; kt < 8; ++kt) {    // h-part, K = 256
      int ko = kt * 32 + kb;
      ar  = wmma_bf16(hfr[kt], *(const v16bf*)(whh + (size_t)(gr + ml) * Hv + ko), ar);
      az  = wmma_bf16(hfr[kt], *(const v16bf*)(whh + (size_t)(gz + ml) * Hv + ko), az);
      anh = wmma_bf16(hfr[kt], *(const v16bf*)(whh + (size_t)(gn + ml) * Hv + ko), anh);
    }

    // ---- elementwise GRU on the C/D layout (reg r -> row mb+r, col ml) ----
    const int hcol = hc + ml;
    const float br  = bi[hcol]          + bh[hcol];
    const float bz  = bi[Hv + hcol]     + bh[Hv + hcol];
    const float bin = bi[2 * Hv + hcol];
    const float bhn = bh[2 * Hv + hcol];
    #pragma unroll
    for (int r = 0; r < 8; ++r) {
      int m = mb + r;
      float rr  = 1.f / (1.f + __expf(-(ar[r] + br)));
      float zz  = 1.f / (1.f + __expf(-(az[r] + bz)));
      float pre = anx[r] + bin + rr * (anh[r] + bhn);
      float e2  = __expf(-2.f * pre);
      float nn  = (1.f - e2) / (1.f + e2);          // tanh
      float hnew = (1.f - zz) * nn + zz * hold[r];
      int bg = b0 + (m >> 2), a = m & 3;
      out[(size_t)((bg * Tv + t) * Av + a) * Hv + hcol] = hnew;
      if (t == Tv - 1) hn_out[(size_t)(bg * Av + a) * Hv + hcol] = hnew;
      hf32[m * HS + hcol] = hnew;
      hb  [m * HS + hcol] = (bf16_t)hnew;
    }
  }
}

// ---------------------------------------------------------------------------
extern "C" void kernel_launch(void* const* d_in, const int* in_sizes, int n_in,
                              void* d_out, int out_size, void* d_ws, size_t ws_size,
                              hipStream_t stream) {
  (void)in_sizes; (void)n_in; (void)out_size; (void)ws_size;
  const float* x       = (const float*)d_in[0];
  const int*   is_init = (const int*)d_in[1];   // bool per harness integer rule
  const float* W_ih    = (const float*)d_in[2];
  const float* W_hh    = (const float*)d_in[3];
  const float* b_ih    = (const float*)d_in[4];
  const float* b_hh    = (const float*)d_in[5];
  float* out = (float*)d_out;

  bf16_t* wih = (bf16_t*)d_ws;                                             // 192 KB
  bf16_t* whh = (bf16_t*)((char*)d_ws + (size_t)G3 * Iv * sizeof(bf16_t)); // 384 KB

  hipLaunchKernelGGL(cvt_weights, dim3((G3 * Hv + 255) / 256), dim3(256), 0, stream,
                     W_ih, W_hh, wih, whh);
  hipLaunchKernelGGL(gru_scan, dim3(Bv / 4), dim3(512), 0, stream,
                     x, is_init, wih, whh, b_ih, b_hh, out);
}
